// GPTMini_85091892069042
// MI455X (gfx1250) — matrix operand order repaired and checked by reading, not compiled, not run on hardware
//
#include <hip/hip_runtime.h>
#include <hip/hip_bf16.h>
#include <math.h>

// ---------------- model dims ----------------
#define GPT_V  32000
#define GPT_D  1024
#define GPT_L  6
#define GPT_H  16
#define GPT_B  4
#define GPT_T  1024
#define GPT_FF 4096
#define GPT_HD 64
#define GPT_M  (GPT_B*GPT_T)   // 4096 token rows

typedef __bf16 bf16_t;
typedef __attribute__((ext_vector_type(16))) __bf16 v16bf;
typedef __attribute__((ext_vector_type(8)))  __bf16 v8bf;
typedef __attribute__((ext_vector_type(8)))  float  v8f;

// ---------------- WMMA helpers ----------------
__device__ __forceinline__ v8bf ldg8(const bf16_t* p) { return *(const v8bf*)p; }

__device__ __forceinline__ v16bf cat16(v8bf lo, v8bf hi) {
  return __builtin_shufflevector(lo, hi, 0,1,2,3,4,5,6,7,8,9,10,11,12,13,14,15);
}

// A 16x32 bf16 fragment: lane (h*16+m) holds K = h*16..h*16+15
__device__ __forceinline__ v16bf load_a_frag(const bf16_t* rowk0, int h) {
  return cat16(ldg8(rowk0 + h*16), ldg8(rowk0 + h*16 + 8));
}
// B 32x16 bf16 fragment: lane (h*16+n) holds K = h*16..h*16+15 (caller applies +h*16)
__device__ __forceinline__ v16bf load_b_frag(const bf16_t* colk) {
  return cat16(ldg8(colk), ldg8(colk + 8));
}
__device__ __forceinline__ v8f wmma_bf16(v16bf a, v16bf b, v8f c) {
  return __builtin_amdgcn_wmma_f32_16x16x32_bf16(false, a, false, b, (short)0, c,
                                                 false, false);
}

// issue one 16-byte global -> LDS async copy (per active lane)
__device__ __forceinline__ void async_copy16(unsigned lds_off, const bf16_t* gp) {
  asm volatile("global_load_async_to_lds_b128 %0, %1, off"
               :: "v"(lds_off), "v"(gp) : "memory");
}

// ---------------- GEMM: C[M,N] = A[M,K] * W[N,K]^T  (bf16 in, f32 out) ----------------
// grid = (N/256, M/64), block = 128 (4 waves); wave w owns N slice [w*64, w*64+64).
// A tile (64x32 bf16, 4KB) is staged in LDS with async global->LDS copies,
// double-buffered so the next tile's fill overlaps this tile's WMMAs.
__global__ __launch_bounds__(128) void gemm_kernel(
    const bf16_t* __restrict__ A, int lda,
    const bf16_t* __restrict__ Bw, int ldb,
    float* __restrict__ C, int ldc, int K)
{
  const int tid  = threadIdx.x;
  const int lane = tid & 31;
  const int wave = tid >> 5;
  const int h    = lane >> 4;
  const int sub  = lane & 15;
  const long mBase = (long)blockIdx.y * 64;
  const long nBase = (long)blockIdx.x * 256 + wave * 64;

  __shared__ bf16_t asA[2][64 * 32];   // double-buffered A tile

  // each thread owns 32 contiguous bytes of the tile: row = tid/2, col half = tid&1
  const int arow = tid >> 1;
  const int acol = (tid & 1) * 16;               // elements
  const bf16_t* agsrc = A + (mBase + arow) * (long)lda + acol;
  const unsigned aoff = (unsigned)(arow * 32 + acol);   // element offset in tile
  const unsigned ab0 = (unsigned)(size_t)&asA[0][aoff];
  const unsigned ab1 = (unsigned)(size_t)&asA[1][aoff];

  v8f acc[4][4];
  #pragma unroll
  for (int mt = 0; mt < 4; ++mt)
    #pragma unroll
    for (int nt = 0; nt < 4; ++nt) acc[mt][nt] = (v8f)0.0f;

  // preload k-tile 0 into buffer 0 (2 chunks of 16B per thread)
  async_copy16(ab0,     agsrc);
  async_copy16(ab0 + 16, agsrc + 8);

  for (int k0 = 0; k0 < K; k0 += 32) {
    const int cur = (k0 >> 5) & 1;
    if (k0 + 32 < K) {   // prefetch next k-tile into the other buffer
      const unsigned nb = cur ? ab0 : ab1;
      async_copy16(nb,      agsrc + k0 + 32);
      async_copy16(nb + 16, agsrc + k0 + 40);
      asm volatile("s_wait_asynccnt 0x2" ::: "memory");  // oldest buffer complete
    } else {
      asm volatile("s_wait_asynccnt 0x0" ::: "memory");
    }
    __syncthreads();

    const bf16_t* as = asA[cur];
    v16bf af[4];
    #pragma unroll
    for (int mt = 0; mt < 4; ++mt) {
      const bf16_t* ar = &as[(mt * 16 + sub) * 32];
      af[mt] = cat16(*(const v8bf*)&ar[h * 16], *(const v8bf*)&ar[h * 16 + 8]);
    }
    #pragma unroll
    for (int nt = 0; nt < 4; ++nt) {
      const bf16_t* bp = Bw + (nBase + nt * 16 + sub) * (long)ldb + k0 + h * 16;
      __builtin_prefetch((const void*)(bp + 64), 0, 0);
      v16bf bfr = load_b_frag(bp);
      #pragma unroll
      for (int mt = 0; mt < 4; ++mt)
        acc[mt][nt] = wmma_bf16(af[mt], bfr, acc[mt][nt]);
    }
    __syncthreads();   // all waves done reading before buffer is overwritten
  }

  // C/D layout: VGPR r, lanes 0-15 -> row r ; lanes 16-31 -> row r+8 ; col = lane&15
  #pragma unroll
  for (int mt = 0; mt < 4; ++mt)
    #pragma unroll
    for (int nt = 0; nt < 4; ++nt)
      #pragma unroll
      for (int r = 0; r < 8; ++r) {
        long m = mBase + mt * 16 + r + 8 * h;
        long n = nBase + nt * 16 + sub;
        C[m * (long)ldc + n] = acc[mt][nt][r];
      }
}

// ---------------- flash attention: one wave per (b, h, 16-query tile) ----------------
__global__ __launch_bounds__(32) void attn_kernel(
    const bf16_t* __restrict__ qkv,  // [B,T,3D] (q | k | v)
    const bf16_t* __restrict__ vT,   // [B,H,HD,T]
    bf16_t* __restrict__ y)          // [B,T,D]
{
  const int b  = blockIdx.x / GPT_H;
  const int hh = blockIdx.x % GPT_H;
  const int q0 = blockIdx.y * 16;
  const int lane = threadIdx.x;
  const int h = lane >> 4, sub = lane & 15;

  const bf16_t* qbase = qkv + ((long)b * GPT_T) * (3 * GPT_D);
  const bf16_t* vtb   = vT + ((long)(b * GPT_H + hh)) * GPT_HD * GPT_T;
  const bf16_t* qrow  = qbase + (long)(q0 + sub) * (3 * GPT_D) + hh * GPT_HD;

  v8f o0 = (v8f)0.0f, o1 = (v8f)0.0f, o2 = (v8f)0.0f, o3 = (v8f)0.0f;
  float mrow[8], lrow[8];
  #pragma unroll
  for (int r = 0; r < 8; ++r) { mrow[r] = -3.0e38f; lrow[r] = 0.0f; }

  __shared__ bf16_t pls[16 * 32];   // P tile, row-major 16x32

  for (int k0 = 0; k0 <= q0 + 15; k0 += 32) {
    // ---- S = Q * K^T for 32 keys (two 16-col accumulators) ----
    v8f s0 = (v8f)0.0f, s1 = (v8f)0.0f;
    #pragma unroll
    for (int dc = 0; dc < GPT_HD; dc += 32) {
      v16bf aq = load_a_frag(qrow + dc, h);
      const bf16_t* kp0 = qbase + (long)(k0 + sub)      * (3*GPT_D) + GPT_D + hh*GPT_HD + dc + h*16;
      const bf16_t* kp1 = qbase + (long)(k0 + 16 + sub) * (3*GPT_D) + GPT_D + hh*GPT_HD + dc + h*16;
      s0 = wmma_bf16(aq, load_b_frag(kp0), s0);
      s1 = wmma_bf16(aq, load_b_frag(kp1), s1);
    }
    // ---- scale + causal mask ----
    #pragma unroll
    for (int r = 0; r < 8; ++r) {
      int qg = q0 + r + 8*h;
      float v0 = s0[r] * 0.125f;           // 1/sqrt(64)
      float v1 = s1[r] * 0.125f;
      s0[r] = (k0 + sub      <= qg) ? v0 : -1.0e30f;
      s1[r] = (k0 + 16 + sub <= qg) ? v1 : -1.0e30f;
    }
    // ---- online softmax (row stats replicated across each 16-lane half) ----
    #pragma unroll
    for (int r = 0; r < 8; ++r) {
      float mx = fmaxf(s0[r], s1[r]);
      #pragma unroll
      for (int off = 1; off < 16; off <<= 1) mx = fmaxf(mx, __shfl_xor(mx, off, 32));
      float mnew  = fmaxf(mrow[r], mx);
      float alpha = __expf(mrow[r] - mnew);
      mrow[r] = mnew;
      float p0 = __expf(s0[r] - mnew);
      float p1 = __expf(s1[r] - mnew);
      int m = r + 8*h;
      pls[m*32 + sub]      = (bf16_t)p0;
      pls[m*32 + 16 + sub] = (bf16_t)p1;
      float rs = p0 + p1;
      #pragma unroll
      for (int off = 1; off < 16; off <<= 1) rs += __shfl_xor(rs, off, 32);
      lrow[r] = lrow[r] * alpha + rs;
      o0[r] *= alpha; o1[r] *= alpha; o2[r] *= alpha; o3[r] *= alpha;
    }
    __syncthreads();
    // ---- O += P * V  (P re-laid-out via LDS into an A fragment) ----
    v16bf ap = cat16(ldg8(&pls[sub*32 + h*16]), ldg8(&pls[sub*32 + h*16 + 8]));
    const bf16_t* vrow = vtb + (long)sub * GPT_T + k0 + h*16;
    o0 = wmma_bf16(ap, load_b_frag(vrow + 0L*16*GPT_T), o0);
    o1 = wmma_bf16(ap, load_b_frag(vrow + 1L*16*GPT_T), o1);
    o2 = wmma_bf16(ap, load_b_frag(vrow + 2L*16*GPT_T), o2);
    o3 = wmma_bf16(ap, load_b_frag(vrow + 3L*16*GPT_T), o3);
    __syncthreads();
  }
  // ---- normalize + store ----
  #pragma unroll
  for (int r = 0; r < 8; ++r) {
    long trow = (long)b * GPT_T + q0 + r + 8*h;
    bf16_t* yr = y + trow * GPT_D + hh * GPT_HD + sub;
    float inv = 1.0f / lrow[r];
    yr[0]  = (bf16_t)(o0[r] * inv);
    yr[16] = (bf16_t)(o1[r] * inv);
    yr[32] = (bf16_t)(o2[r] * inv);
    yr[48] = (bf16_t)(o3[r] * inv);
  }
}

// ---------------- elementwise / normalization kernels ----------------
__global__ __launch_bounds__(256) void embed_kernel(
    const int* __restrict__ idx, const float* __restrict__ tok,
    const float* __restrict__ pos, float* __restrict__ xf)
{
  long i = (long)blockIdx.x * blockDim.x + threadIdx.x;
  if (i >= (long)GPT_M * GPT_D) return;
  long bt = i / GPT_D; int d = (int)(i % GPT_D); int t = (int)(bt % GPT_T);
  xf[i] = tok[(long)idx[bt] * GPT_D + d] + pos[(long)t * GPT_D + d];
}

__global__ __launch_bounds__(256) void ln_kernel(
    const float* __restrict__ x, const float* __restrict__ g,
    const float* __restrict__ b, bf16_t* __restrict__ out, int Dd)
{
  long row = blockIdx.x;
  const float* xr = x + row * (long)Dd;
  float s = 0.0f, s2 = 0.0f;
  for (int i = threadIdx.x; i < Dd; i += 256) { float v = xr[i]; s += v; s2 += v*v; }
  __shared__ float rs[256], rs2[256];
  rs[threadIdx.x] = s; rs2[threadIdx.x] = s2; __syncthreads();
  for (int off = 128; off > 0; off >>= 1) {
    if ((int)threadIdx.x < off) {
      rs[threadIdx.x]  += rs[threadIdx.x + off];
      rs2[threadIdx.x] += rs2[threadIdx.x + off];
    }
    __syncthreads();
  }
  float mu  = rs[0] / (float)Dd;
  float var = rs2[0] / (float)Dd - mu * mu;
  float inv = rsqrtf(var + 1e-5f);
  for (int i = threadIdx.x; i < Dd; i += 256)
    out[row * (long)Dd + i] = (bf16_t)((xr[i] - mu) * inv * g[i] + b[i]);
}

__global__ __launch_bounds__(256) void cast_kernel(
    const float* __restrict__ in, bf16_t* __restrict__ out, long n)
{
  long i = (long)blockIdx.x * blockDim.x + threadIdx.x;
  if (i < n) out[i] = (bf16_t)in[i];
}

__global__ __launch_bounds__(256) void gelu_kernel(
    const float* __restrict__ in, const float* __restrict__ bias,
    bf16_t* __restrict__ out, long n, int nf)
{
  long i = (long)blockIdx.x * blockDim.x + threadIdx.x;
  if (i >= n) return;
  float v = in[i] + bias[i % nf];
  out[i] = (bf16_t)(0.5f * v * (1.0f + erff(v * 0.70710678f)));
}

__global__ __launch_bounds__(256) void resid_kernel(
    float* __restrict__ x, const float* __restrict__ t,
    const float* __restrict__ bias, long n, int nf)
{
  long i = (long)blockIdx.x * blockDim.x + threadIdx.x;
  if (i >= n) return;
  float bv = bias ? bias[i % nf] : 0.0f;
  x[i] = x[i] + t[i] + bv;
}

__global__ __launch_bounds__(256) void transpose_v_kernel(
    const bf16_t* __restrict__ qkv, bf16_t* __restrict__ vT)
{
  long i = (long)blockIdx.x * blockDim.x + threadIdx.x;
  if (i >= (long)GPT_M * GPT_D) return;
  long b = i / ((long)GPT_T * GPT_D);
  long rem = i % ((long)GPT_T * GPT_D);
  int t = (int)(rem / GPT_D); int d = (int)(rem % GPT_D);
  int hh = d >> 6; int dd = d & 63;
  vT[((long)(b * GPT_H + hh) * GPT_HD + dd) * GPT_T + t] =
      qkv[((long)b * GPT_T + t) * (3 * GPT_D) + 2 * GPT_D + d];
}

// ---------------- host orchestration ----------------
extern "C" void kernel_launch(void* const* d_in, const int* in_sizes, int n_in,
                              void* d_out, int out_size, void* d_ws, size_t ws_size,
                              hipStream_t stream)
{
  (void)in_sizes; (void)n_in; (void)out_size;
  const int*   idx   = (const int*)  d_in[0];
  const float* tok   = (const float*)d_in[1];
  const float* pos   = (const float*)d_in[2];
  const float* ln1g  = (const float*)d_in[3];
  const float* ln1b  = (const float*)d_in[4];
  const float* qkvw  = (const float*)d_in[5];
  const float* projw = (const float*)d_in[6];
  const float* ln2g  = (const float*)d_in[7];
  const float* ln2b  = (const float*)d_in[8];
  const float* w1    = (const float*)d_in[9];
  const float* b1    = (const float*)d_in[10];
  const float* w2    = (const float*)d_in[11];
  const float* b2    = (const float*)d_in[12];
  const float* lnfg  = (const float*)d_in[13];
  const float* lnfb  = (const float*)d_in[14];
  const float* headw = (const float*)d_in[15];
  float* out = (float*)d_out;

  char* p = (char*)d_ws;
  auto take = [&](size_t bytes) -> char* {
    char* r = p; p += (bytes + 255) & ~(size_t)255; return r;
  };
  float*  xf    = (float*) take((size_t)GPT_M * GPT_D * 4);
  float*  tmp   = (float*) take((size_t)GPT_M * GPT_FF * 4);
  bf16_t* hb    = (bf16_t*)take((size_t)GPT_M * GPT_FF * 2);
  bf16_t* qkvb  = (bf16_t*)take((size_t)GPT_M * 3 * GPT_D * 2);
  bf16_t* vtb   = (bf16_t*)take((size_t)GPT_M * GPT_D * 2);
  bf16_t* yb    = (bf16_t*)take((size_t)GPT_M * GPT_D * 2);
  bf16_t* wb    = (bf16_t*)take((size_t)8 * GPT_D * GPT_D * 2);
  bf16_t* headb = (bf16_t*)take((size_t)GPT_V * GPT_D * 2);
  if ((size_t)(p - (char*)d_ws) > ws_size) return;

  const long nMD = (long)GPT_M * GPT_D;
  const long nMF = (long)GPT_M * GPT_FF;
  dim3 b256(256), b128(128), b32(32);
  auto g1 = [](long n) { return dim3((unsigned)((n + 255) / 256)); };

  embed_kernel<<<g1(nMD), b256, 0, stream>>>(idx, tok, pos, xf);

  for (int l = 0; l < GPT_L; ++l) {
    // --- attention block ---
    ln_kernel<<<dim3(GPT_M), b256, 0, stream>>>(xf, ln1g + (long)l*GPT_D,
                                                ln1b + (long)l*GPT_D, hb, GPT_D);
    long nqkv = 3L * GPT_D * GPT_D;
    cast_kernel<<<g1(nqkv), b256, 0, stream>>>(qkvw + (long)l*nqkv, wb, nqkv);
    gemm_kernel<<<dim3(3*GPT_D/256, GPT_M/64), b128, 0, stream>>>(
        hb, GPT_D, wb, GPT_D, tmp, 3*GPT_D, GPT_D);
    long n3 = (long)GPT_M * 3 * GPT_D;
    cast_kernel<<<g1(n3), b256, 0, stream>>>(tmp, qkvb, n3);
    transpose_v_kernel<<<g1(nMD), b256, 0, stream>>>(qkvb, vtb);
    attn_kernel<<<dim3(GPT_B*GPT_H, GPT_T/16), b32, 0, stream>>>(qkvb, vtb, yb);

    long nDD = (long)GPT_D * GPT_D;
    cast_kernel<<<g1(nDD), b256, 0, stream>>>(projw + (long)l*nDD, wb, nDD);
    gemm_kernel<<<dim3(GPT_D/256, GPT_M/64), b128, 0, stream>>>(
        yb, GPT_D, wb, GPT_D, tmp, GPT_D, GPT_D);
    resid_kernel<<<g1(nMD), b256, 0, stream>>>(xf, tmp, (const float*)nullptr,
                                               nMD, GPT_D);
    // --- MLP block ---
    ln_kernel<<<dim3(GPT_M), b256, 0, stream>>>(xf, ln2g + (long)l*GPT_D,
                                                ln2b + (long)l*GPT_D, hb, GPT_D);
    long nFD = (long)GPT_FF * GPT_D;
    cast_kernel<<<g1(nFD), b256, 0, stream>>>(w1 + (long)l*nFD, wb, nFD);
    gemm_kernel<<<dim3(GPT_FF/256, GPT_M/64), b128, 0, stream>>>(
        hb, GPT_D, wb, GPT_D, tmp, GPT_FF, GPT_D);
    gelu_kernel<<<g1(nMF), b256, 0, stream>>>(tmp, b1 + (long)l*GPT_FF, hb,
                                              nMF, GPT_FF);
    cast_kernel<<<g1(nFD), b256, 0, stream>>>(w2 + (long)l*nFD, wb, nFD);
    gemm_kernel<<<dim3(GPT_D/256, GPT_M/64), b128, 0, stream>>>(
        hb, GPT_FF, wb, GPT_FF, tmp, GPT_D, GPT_FF);
    resid_kernel<<<g1(nMD), b256, 0, stream>>>(xf, tmp, b2 + (long)l*GPT_D,
                                               nMD, GPT_D);
  }

  // --- final LN + logits ---
  ln_kernel<<<dim3(GPT_M), b256, 0, stream>>>(xf, lnfg, lnfb, hb, GPT_D);
  long nVD = (long)GPT_V * GPT_D;
  cast_kernel<<<g1(nVD), b256, 0, stream>>>(headw, headb, nVD);
  gemm_kernel<<<dim3(GPT_V/256, GPT_M/64), b128, 0, stream>>>(
      hb, GPT_D, headb, GPT_D, out, GPT_V, GPT_D);
}
